// GATRecommender_78692390797382
// MI455X (gfx1250) — matrix operand order repaired and checked by reading, not compiled, not run on hardware
//
#include <hip/hip_runtime.h>
#include <hip/hip_bf16.h>

// ---------------------------------------------------------------------------
// GATRecommender forward, CDNA5 (gfx1250, wave32).
// Dense GEMMs -> v_wmma_f32_16x16x32_bf16, bf16 operands pre-converted.
// One wave owns a 32x64 output tile: per K-step 12 contiguous b128 loads feed
// 8 WMMAs (each B fragment reused by 2 WMMAs, each A fragment by 4).
// Sparse GAT softmax/aggregation -> wave32 VALU + global atomics.
// ---------------------------------------------------------------------------

#define NUM_USERS 1024
#define N_NODES   3072
#define HID       768
#define HEADS     8

typedef __attribute__((ext_vector_type(16))) __bf16 v16bf;
typedef __attribute__((ext_vector_type(8)))  __bf16 v8bf;
typedef __attribute__((ext_vector_type(8)))  float  v8f;

static __device__ __forceinline__ unsigned short bf_bits(float f) {
  unsigned u = __float_as_uint(f);
  unsigned r = u + 0x7FFFu + ((u >> 16) & 1u);           // round-to-nearest-even
  return (unsigned short)(r >> 16);
}
static __device__ __forceinline__ __bf16 f2bf(float f) {
  unsigned short h = bf_bits(f);
  return __builtin_bit_cast(__bf16, h);
}

static __device__ __forceinline__ unsigned enc_ord(float f) {
  unsigned b = __float_as_uint(f);
  return (b & 0x80000000u) ? ~b : (b | 0x80000000u);     // order-preserving map
}
static __device__ __forceinline__ float dec_ord(unsigned u) {
  unsigned b = (u & 0x80000000u) ? (u ^ 0x80000000u) : ~u;
  return __uint_as_float(b);
}

static __device__ __forceinline__ float wave_sum(float v) {
  #pragma unroll
  for (int off = 16; off > 0; off >>= 1) v += __shfl_down(v, off, 32);
  return v;
}

#define CAT16(lo, hi) __builtin_shufflevector(lo, hi, 0, 1, 2, 3, 4, 5, 6, 7, \
                                              8, 9, 10, 11, 12, 13, 14, 15)

// ---------------- one-shot fp32 -> bf16 conversion kernels -----------------
// Elementwise: n must be a multiple of 4. float4 in -> 8B packed bf16 out.
__global__ void gat_cvt_bf16(const float* __restrict__ in, __bf16* __restrict__ out,
                             size_t n4) {
  size_t i = (size_t)blockIdx.x * blockDim.x + threadIdx.x;
  if (i >= n4) return;
  float4 v = ((const float4*)in)[i];
  uint2 o;
  o.x = (unsigned)bf_bits(v.x) | ((unsigned)bf_bits(v.y) << 16);
  o.y = (unsigned)bf_bits(v.z) | ((unsigned)bf_bits(v.w) << 16);
  ((uint2*)out)[i] = o;
}

// Convert + transpose: BT[n*K + k] = bf16(B[k*N + n])  (coalesced writes)
__global__ void gat_cvtT_bf16(const float* __restrict__ B, __bf16* __restrict__ BT,
                              int K, int N) {
  size_t i = (size_t)blockIdx.x * blockDim.x + threadIdx.x;
  if (i >= (size_t)K * N) return;
  int n = (int)(i / K), k = (int)(i - (size_t)n * K);
  BT[i] = f2bf(B[(size_t)k * N + n]);
}

// ------------------------- WMMA GEMM ---------------------------------------
// C[M,N] = A[M,K] * B[K,N] (+bias)(+relu), A bf16 row-major, BT bf16 [N,K].
// One wave computes a 32(M) x 64(N) tile: per K-step 4 A-loads + 8 B-loads
// (all contiguous b128) feed 8 v_wmma_f32_16x16x32_bf16. 4 waves / block.
// Requires M%32==0, N%64==0, K%32==0.
__global__ void gat_gemm_bf16_wmma(const __bf16* __restrict__ A,
                                   const __bf16* __restrict__ BT,
                                   const float* __restrict__ bias,
                                   float* __restrict__ C,
                                   int M, int N, int K, int relu) {
  const int wave = threadIdx.x >> 5;
  const int lane = threadIdx.x & 31;
  const int strips_n = N >> 6;
  const int id = blockIdx.x * 4 + wave;
  if (id >= (M >> 5) * strips_n) return;                 // uniform per wave
  const int tm = id / strips_n;
  const int tn = id - tm * strips_n;
  const int m0 = tm << 5, n0 = tn << 6;
  const int half = lane >> 4;                            // lane-group select
  const int r    = lane & 15;                            // row (A) / col (B)

  // CDNA5 16-bit A(16x32) layout: lane(m=r, group=half) holds
  //   K = kk + 8*half + [0..7]  and  K = kk + 16 + 8*half + [0..7]
  const __bf16* Ap0 = A + (size_t)(m0 + r) * K + 8 * half;        // rows m0..m0+15
  const __bf16* Ap1 = Ap0 + (size_t)16 * K;                       // rows m0+16..m0+31
  // B(32x16) layout: lane(n=r, group=half) holds K = kk + 8*half + [0..7]
  // and K = kk + 16 + 8*half + [0..7]; from BT[n, :].
  const __bf16* Bp = BT + (size_t)(n0 + r) * K + 8 * half;
  const size_t bcol = (size_t)16 * K;                    // 16 columns ahead

  v8f acc[2][4] = {};
  for (int kk = 0; kk < K; kk += 32) {
    v16bf a0 = CAT16(*(const v8bf*)(Ap0 + kk), *(const v8bf*)(Ap0 + kk + 16));
    v16bf a1 = CAT16(*(const v8bf*)(Ap1 + kk), *(const v8bf*)(Ap1 + kk + 16));
    #pragma unroll
    for (int t = 0; t < 4; ++t) {
      const __bf16* bp = Bp + (size_t)t * bcol + kk;
      v16bf bv = CAT16(*(const v8bf*)(bp), *(const v8bf*)(bp + 16));
      acc[0][t] = __builtin_amdgcn_wmma_f32_16x16x32_bf16(
          false, a0, false, bv, (short)0, acc[0][t], false, false);
      acc[1][t] = __builtin_amdgcn_wmma_f32_16x16x32_bf16(
          false, a1, false, bv, (short)0, acc[1][t], false, false);
    }
  }

  // D layout: lanes 0-15 -> M=v, N=lane ; lanes 16-31 -> M=8+v, N=lane-16
  #pragma unroll
  for (int mi = 0; mi < 2; ++mi) {
    #pragma unroll
    for (int t = 0; t < 4; ++t) {
      const int n = n0 + 16 * t + r;
      const float bv2 = bias ? bias[n] : 0.f;
      #pragma unroll
      for (int v = 0; v < 8; ++v) {
        float val = acc[mi][t][v] + bv2;
        if (relu) val = fmaxf(val, 0.f);
        C[(size_t)(m0 + 16 * mi + v + 8 * half) * N + n] = val;
      }
    }
  }
}

// ------------------------- helper kernels ----------------------------------
__global__ void gat_zero(float* p, size_t n) {
  size_t i = (size_t)blockIdx.x * blockDim.x + threadIdx.x;
  size_t stride = (size_t)gridDim.x * blockDim.x;
  for (; i < n; i += stride) p[i] = 0.f;
}

__global__ void gat_img_mean(const float* __restrict__ img, float* __restrict__ out, int B) {
  int i = blockIdx.x * blockDim.x + threadIdx.x;
  if (i >= B * HID) return;
  int b = i / HID, c = i - b * HID;
  const float* p = img + (size_t)b * 3 * HID + c;
  out[i] = (p[0] + p[HID] + p[2 * HID]) * (1.f / 3.f);
}

__global__ void gat_meta_fc(const float* __restrict__ bf, const float* __restrict__ W,
                            const float* __restrict__ b, float* __restrict__ out, int B) {
  int i = blockIdx.x * blockDim.x + threadIdx.x;
  if (i >= B * HID) return;
  int row = i / HID, c = i - row * HID;
  const float* f = bf + row * 3;
  out[i] = f[0] * W[c] + f[1] * W[HID + c] + f[2] * W[2 * HID + c] + b[c];
}

__global__ void gat_scatter_x(const int* __restrict__ uidx, const int* __restrict__ bidx,
                              const float* __restrict__ utab, const float* __restrict__ btab,
                              const float* __restrict__ text, const float* __restrict__ img,
                              const float* __restrict__ meta, float* __restrict__ x, int B) {
  int i = blockIdx.x * blockDim.x + threadIdx.x;
  if (i >= B * HID) return;
  int row = i / HID, c = i - row * HID;
  int u = uidx[row];
  x[(size_t)u * HID + c] = utab[(size_t)u * HID + c];
  int bz = bidx[row];
  float v = 0.25f * (text[i] + img[i] + meta[i] + btab[(size_t)(bz - NUM_USERS) * HID + c]);
  x[(size_t)bz * HID + c] = v;
}

// s[n,h] = sum_c h[n,h,c]*att[h,c] : one wave per (node,head)
__global__ void gat_att_score(const float* __restrict__ h, const float* __restrict__ att,
                              float* __restrict__ s, int N, int heads, int C) {
  int wid = (int)((blockIdx.x * blockDim.x + threadIdx.x) >> 5);
  int lane = threadIdx.x & 31;
  if (wid >= N * heads) return;
  int n = wid / heads, hd = wid - n * heads;
  const float* hp = h + ((size_t)n * heads + hd) * C;
  const float* ap = att + (size_t)hd * C;
  float sum = 0.f;
  for (int c = lane; c < C; c += 32) sum += hp[c] * ap[c];
  sum = wave_sum(sum);
  if (lane == 0) s[wid] = sum;
}

static __device__ __forceinline__ void edge_sd(const int* ei, int e, int E, int& s, int& d) {
  if (e < E) { s = ei[e]; d = ei[E + e]; } else { s = e - E; d = e - E; }  // self loops
}

__global__ void gat_edge_max(const int* __restrict__ ei, const float* __restrict__ ssrc,
                             const float* __restrict__ sdst, unsigned* __restrict__ emax,
                             int E, int Etot, int heads) {
  int idx = blockIdx.x * blockDim.x + threadIdx.x;
  if (idx >= Etot * heads) return;
  int e = idx / heads, hd = idx - e * heads;
  int s, d; edge_sd(ei, e, E, s, d);
  float v = ssrc[s * heads + hd] + sdst[d * heads + hd];
  v = (v > 0.f) ? v : 0.2f * v;                          // leaky_relu(0.2)
  atomicMax(&emax[d * heads + hd], enc_ord(v));
}

__global__ void gat_edge_exp(const int* __restrict__ ei, const float* __restrict__ ssrc,
                             const float* __restrict__ sdst, const unsigned* __restrict__ emax,
                             float* __restrict__ ee, float* __restrict__ denom,
                             int E, int Etot, int heads) {
  int idx = blockIdx.x * blockDim.x + threadIdx.x;
  if (idx >= Etot * heads) return;
  int e = idx / heads, hd = idx - e * heads;
  int s, d; edge_sd(ei, e, E, s, d);
  float v = ssrc[s * heads + hd] + sdst[d * heads + hd];
  v = (v > 0.f) ? v : 0.2f * v;
  float x = expf(v - dec_ord(emax[d * heads + hd]));
  ee[idx] = x;
  atomicAdd(&denom[d * heads + hd], x);
}

// one wave per (edge, head): agg[d,hd,:] += h[s,hd,:] * alpha
__global__ void gat_aggregate(const int* __restrict__ ei, const float* __restrict__ h,
                              const float* __restrict__ ee, const float* __restrict__ denom,
                              float* __restrict__ agg, int E, int Etot, int heads, int C) {
  int wid = (int)((blockIdx.x * blockDim.x + threadIdx.x) >> 5);
  int lane = threadIdx.x & 31;
  if (wid >= Etot * heads) return;
  int e = wid / heads, hd = wid - e * heads;
  int s, d; edge_sd(ei, e, E, s, d);
  float alpha = ee[wid] / (denom[d * heads + hd] + 1e-16f);
  const float* hp = h + ((size_t)s * heads + hd) * C;
  float* op = agg + ((size_t)d * heads + hd) * C;
  for (int c = lane; c < C; c += 32) atomicAdd(&op[c], hp[c] * alpha);
}

__global__ void gat_bias_act(float* __restrict__ buf, const float* __restrict__ bias,
                             size_t total, int cols, int relu) {
  size_t i = (size_t)blockIdx.x * blockDim.x + threadIdx.x;
  if (i >= total) return;
  float v = buf[i] + bias[i % cols];
  if (relu) v = fmaxf(v, 0.f);
  buf[i] = v;
}

__global__ void gat_build_cat(const float* __restrict__ xo, const float* __restrict__ text,
                              const float* __restrict__ img, const int* __restrict__ uidx,
                              const int* __restrict__ bidx, float* __restrict__ cat, int B) {
  int i = blockIdx.x * blockDim.x + threadIdx.x;
  if (i >= B * 4 * HID) return;
  int row = i / (4 * HID), c = i - row * (4 * HID);
  float v;
  if (c < HID)            v = xo[(size_t)uidx[row] * HID + c];
  else if (c < 2 * HID)   v = xo[(size_t)bidx[row] * HID + (c - HID)];
  else if (c < 3 * HID)   v = text[(size_t)row * HID + (c - 2 * HID)];
  else                    v = img[(size_t)row * HID + (c - 3 * HID)];
  cat[i] = v;
}

// out[i] = f2[i,:] . Wf3 + bf3 : one wave per row
__global__ void gat_final_dot(const float* __restrict__ f2, const float* __restrict__ Wf3,
                              const float* __restrict__ bf3, float* __restrict__ out, int B) {
  int wid = (int)((blockIdx.x * blockDim.x + threadIdx.x) >> 5);
  int lane = threadIdx.x & 31;
  if (wid >= B) return;
  const float* p = f2 + (size_t)wid * HID;
  float sum = 0.f;
  for (int c = lane; c < HID; c += 32) sum += p[c] * Wf3[c];
  sum = wave_sum(sum);
  if (lane == 0) out[wid] = sum + bf3[0];
}

// ---------------------------------------------------------------------------
static inline int cdiv(long long a, long long b) { return (int)((a + b - 1) / b); }

extern "C" void kernel_launch(void* const* d_in, const int* in_sizes, int n_in,
                              void* d_out, int out_size, void* d_ws, size_t ws_size,
                              hipStream_t stream) {
  const float* text_cls = (const float*)d_in[0];
  const float* img_cls  = (const float*)d_in[1];
  const float* biz_f    = (const float*)d_in[2];
  const float* W_text   = (const float*)d_in[3];
  const float* b_text   = (const float*)d_in[4];
  const float* W_img    = (const float*)d_in[5];
  const float* b_img    = (const float*)d_in[6];
  const float* W_bf     = (const float*)d_in[7];
  const float* b_bf     = (const float*)d_in[8];
  const float* utab     = (const float*)d_in[9];
  const float* btab     = (const float*)d_in[10];
  const float* W1       = (const float*)d_in[11];
  const float* asrc1    = (const float*)d_in[12];
  const float* adst1    = (const float*)d_in[13];
  const float* b1       = (const float*)d_in[14];
  const float* W2       = (const float*)d_in[15];
  const float* asrc2    = (const float*)d_in[16];
  const float* adst2    = (const float*)d_in[17];
  const float* b2       = (const float*)d_in[18];
  const float* Wf1      = (const float*)d_in[19];
  const float* bf1      = (const float*)d_in[20];
  const float* Wf2      = (const float*)d_in[21];
  const float* bf2      = (const float*)d_in[22];
  const float* Wf3      = (const float*)d_in[23];
  const float* bf3      = (const float*)d_in[24];
  const int*   uidx     = (const int*)d_in[25];
  const int*   bidx     = (const int*)d_in[26];
  const int*   eidx     = (const int*)d_in[27];

  const int B    = in_sizes[25];          // 1024
  const int E    = in_sizes[27] / 2;      // 16384
  const int Etot = E + N_NODES;           // + self loops
  float* out = (float*)d_out;

  // ---- workspace carve-up (float units) ----
  float* ws = (float*)d_ws; size_t o = 0;
  float* x     = ws + o; o += (size_t)N_NODES * HID;
  float* temb  = ws + o; o += (size_t)B * HID;
  float* imean = ws + o; o += (size_t)B * HID;
  float* iemb  = ws + o; o += (size_t)B * HID;
  float* meta  = ws + o; o += (size_t)B * HID;
  float* h1    = ws + o; o += (size_t)N_NODES * HEADS * HID;
  float* agg1  = ws + o; o += (size_t)N_NODES * HEADS * HID;
  float* h2    = ws + o; o += (size_t)N_NODES * HID;
  float* agg2  = ws + o; o += (size_t)N_NODES * HID;
  float* ss1   = ws + o; o += (size_t)N_NODES * HEADS;
  float* sd1   = ws + o; o += (size_t)N_NODES * HEADS;
  unsigned* em1 = (unsigned*)(ws + o); o += (size_t)N_NODES * HEADS;
  float* dn1   = ws + o; o += (size_t)N_NODES * HEADS;
  float* ee1   = ws + o; o += (size_t)Etot * HEADS;
  float* ss2   = ws + o; o += N_NODES;
  float* sd2   = ws + o; o += N_NODES;
  unsigned* em2 = (unsigned*)(ws + o); o += N_NODES;
  float* dn2   = ws + o; o += N_NODES;
  float* ee2   = ws + o; o += Etot;
  float* cat   = ws + o; o += (size_t)B * 4 * HID;
  float* f1    = ws + o; o += (size_t)B * 2 * HID;
  float* f2    = ws + o; o += (size_t)B * HID;
  o = (o + 7) & ~(size_t)7;
  __bf16* bfA  = (__bf16*)(ws + o); o += (size_t)N_NODES * HEADS * HID / 2;  // largest A
  __bf16* bfB  = (__bf16*)(ws + o); o += (size_t)HEADS * HID * HID / 2;      // largest B^T

  const int T = 256;
  auto cvt = [&](const float* src, __bf16* dst, size_t n) {
    gat_cvt_bf16<<<cdiv((long long)(n / 4), T), T, 0, stream>>>(src, dst, n / 4);
  };
  auto cvtT = [&](const float* src, __bf16* dst, int K, int N) {
    gat_cvtT_bf16<<<cdiv((long long)K * N, T), T, 0, stream>>>(src, dst, K, N);
  };
  // full GEMM: convert A and B^T to bf16, then WMMA tiles of 32x64
  auto gemm = [&](const float* A, const float* Bm, const float* bias, float* C,
                  int M, int N, int K, int relu) {
    cvt(A, bfA, (size_t)M * K);
    cvtT(Bm, bfB, K, N);
    int tiles = (M >> 5) * (N >> 6);
    gat_gemm_bf16_wmma<<<cdiv(tiles, 4), 128, 0, stream>>>(bfA, bfB, bias, C, M, N, K, relu);
  };

  // ---- encoders ----
  gat_zero<<<1024, T, 0, stream>>>(x, (size_t)N_NODES * HID);
  gat_img_mean<<<cdiv((long long)B * HID, T), T, 0, stream>>>(img_cls, imean, B);
  gemm(text_cls, W_text, b_text, temb, B, HID, HID, 0);
  gemm(imean,    W_img,  b_img,  iemb, B, HID, HID, 0);
  gat_meta_fc<<<cdiv((long long)B * HID, T), T, 0, stream>>>(biz_f, W_bf, b_bf, meta, B);
  gat_scatter_x<<<cdiv((long long)B * HID, T), T, 0, stream>>>(
      uidx, bidx, utab, btab, temb, iemb, meta, x, B);

  // ---- GAT layer 1 (8 heads) ----
  gemm(x, W1, nullptr, h1, N_NODES, HEADS * HID, HID, 0);
  gat_att_score<<<cdiv((long long)N_NODES * HEADS * 32, T), T, 0, stream>>>(h1, asrc1, ss1, N_NODES, HEADS, HID);
  gat_att_score<<<cdiv((long long)N_NODES * HEADS * 32, T), T, 0, stream>>>(h1, adst1, sd1, N_NODES, HEADS, HID);
  gat_zero<<<64, T, 0, stream>>>((float*)em1, (size_t)N_NODES * HEADS);   // 0u == ordered(-inf)
  gat_zero<<<64, T, 0, stream>>>(dn1, (size_t)N_NODES * HEADS);
  gat_edge_max<<<cdiv((long long)Etot * HEADS, T), T, 0, stream>>>(eidx, ss1, sd1, em1, E, Etot, HEADS);
  gat_edge_exp<<<cdiv((long long)Etot * HEADS, T), T, 0, stream>>>(eidx, ss1, sd1, em1, ee1, dn1, E, Etot, HEADS);
  gat_zero<<<2048, T, 0, stream>>>(agg1, (size_t)N_NODES * HEADS * HID);
  gat_aggregate<<<cdiv((long long)Etot * HEADS * 32, T), T, 0, stream>>>(eidx, h1, ee1, dn1, agg1, E, Etot, HEADS, HID);
  gat_bias_act<<<cdiv((long long)N_NODES * HEADS * HID, T), T, 0, stream>>>(
      agg1, b1, (size_t)N_NODES * HEADS * HID, HEADS * HID, 1);

  // ---- GAT layer 2 (1 head) ----
  gemm(agg1, W2, nullptr, h2, N_NODES, HID, HEADS * HID, 0);
  gat_att_score<<<cdiv((long long)N_NODES * 32, T), T, 0, stream>>>(h2, asrc2, ss2, N_NODES, 1, HID);
  gat_att_score<<<cdiv((long long)N_NODES * 32, T), T, 0, stream>>>(h2, adst2, sd2, N_NODES, 1, HID);
  gat_zero<<<16, T, 0, stream>>>((float*)em2, (size_t)N_NODES);
  gat_zero<<<16, T, 0, stream>>>(dn2, (size_t)N_NODES);
  gat_edge_max<<<cdiv(Etot, T), T, 0, stream>>>(eidx, ss2, sd2, em2, E, Etot, 1);
  gat_edge_exp<<<cdiv(Etot, T), T, 0, stream>>>(eidx, ss2, sd2, em2, ee2, dn2, E, Etot, 1);
  gat_zero<<<1024, T, 0, stream>>>(agg2, (size_t)N_NODES * HID);
  gat_aggregate<<<cdiv((long long)Etot * 32, T), T, 0, stream>>>(eidx, h2, ee2, dn2, agg2, E, Etot, 1, HID);
  gat_bias_act<<<cdiv((long long)N_NODES * HID, T), T, 0, stream>>>(
      agg2, b2, (size_t)N_NODES * HID, HID, 0);

  // ---- fusion MLP ----
  gat_build_cat<<<cdiv((long long)B * 4 * HID, T), T, 0, stream>>>(agg2, temb, iemb, uidx, bidx, cat, B);
  gemm(cat, Wf1, bf1, f1, B, 2 * HID, 4 * HID, 1);
  gemm(f1,  Wf2, bf2, f2, B, HID, 2 * HID, 1);
  gat_final_dot<<<cdiv((long long)B * 32, T), T, 0, stream>>>(f2, Wf3, bf3, out, B);
}
